// MixtureOfExperts_33981781246195
// MI455X (gfx1250) — matrix operand order repaired and checked by reading, not compiled, not run on hardware
//
#include <hip/hip_runtime.h>
#include <hip/hip_bf16.h>

// ---------------- problem constants (from reference setup_inputs) ----------
#define E_NUM   8
#define C_DIM   1024
#define H_DIM   4096
#define N_TOK   8192           // B*T = 4*2048
#define M_TILE  64             // tokens per FFN workgroup
#define KCH     32             // H-chunk = WMMA K for bf16

// LDS layout (bytes) for moe_ffn — all operands in WMMA fragment-major form:
// tile = 512 bf16 = 1024 B; element addr = tileBase + lane*32B + elem*2B
#define XS_OFF  0                                   // 128 A-tiles [mtile][ktile]
#define W1_OFF  (XS_OFF + 128 * 1024)               // 64 B-tiles [ktile][ntile]
#define W2_OFF  (W1_OFF + 64 * 1024)                // 64 B-tiles [ntile]
#define HS_OFF  (W2_OFF + 64 * 1024)                // 4 A-tiles (h chunk)
#define HR_OFF  (HS_OFF + 4 * 1024)                 // 8 f32 C-tiles (K-split reduce)
#define TOK_OFF (HR_OFF + 8 * 1024)
#define SMEM_FFN (TOK_OFF + M_TILE * 4)             // 274688 B (< 320KB WGP LDS)

// Workspace layout (bytes): needs ~65 MB
#define WS_CNT   0              // 8 ints
#define WS_BASE  4096           // 8 ints
#define WS_IDX   8192           // E*N ints   (256 KB)
#define WS_EPOS  (WS_IDX + E_NUM * N_TOK * 4)       // N*2 ints (64 KB)
#define WS_TOKW  (WS_EPOS + N_TOK * 2 * 4)          // N*2 floats (64 KB)
#define WS_PART  (1 << 20)      // 2N*C fp32 partial rows (64 MB)

// Scheduler pipelining directives: batch a stage's WMMAs, then batch the
// next stage's LDS reads, so fragment buffers overlap -> distinct registers
// -> non-blocking s_wait_dscnt.
#if defined(__has_builtin)
#if __has_builtin(__builtin_amdgcn_sched_group_barrier)
#define SGB(mask, size) __builtin_amdgcn_sched_group_barrier((mask), (size), 0)
#endif
#endif
#ifndef SGB
#define SGB(mask, size)
#endif
#define SGB_WMMA(n) SGB(0x008, n)   // n matrix ops
#define SGB_DSR(n)  SGB(0x100, n)   // n LDS reads

typedef __attribute__((ext_vector_type(16))) __bf16        v16bf;
typedef __attribute__((ext_vector_type(8)))  float         v8f;
typedef __attribute__((ext_vector_type(8)))  unsigned int  v8u;

// One fragment = 32 contiguous bytes per lane -> 2x ds_load_b128
__device__ __forceinline__ v16bf load_frag(const __bf16* tile, int lane) {
  return __builtin_bit_cast(v16bf, *(const v8u*)(tile + (lane << 4)));
}

// Fragment element maps (ISA 7.12.2, 16-bit layouts):
//  A (16x32): lane = (row&15) + 16*((K>>3)&1); elem = (K&7) + 8*(K>>4)
//  B (32x16): lane = (col&15) + 16*((K>>3)&1); elem = (K&7) + 8*(K>>4)

// ---------------------------------------------------------------------------
// Kernel A: gating. One wave32 per token; top-2 + renormalize.
// ---------------------------------------------------------------------------
__global__ void __launch_bounds__(256) moe_gate(
    const float* __restrict__ x, const float* __restrict__ gate_w,
    int* __restrict__ cnt, int* __restrict__ idx_list,
    int* __restrict__ tok_epos, float* __restrict__ tok_w) {
  __shared__ float gws[E_NUM][C_DIM];            // 32 KB
  for (int i = threadIdx.x; i < E_NUM * C_DIM; i += blockDim.x)
    gws[i >> 10][i & (C_DIM - 1)] = gate_w[i];
  __syncthreads();

  int wave = threadIdx.x >> 5;
  int lane = threadIdx.x & 31;
  int n = blockIdx.x * 8 + wave;
  const float* xr = x + (size_t)n * C_DIM;

  float acc[E_NUM];
#pragma unroll
  for (int e = 0; e < E_NUM; ++e) acc[e] = 0.f;
  for (int c = lane; c < C_DIM; c += 32) {
    float xv = xr[c];
#pragma unroll
    for (int e = 0; e < E_NUM; ++e) acc[e] += xv * gws[e][c];
  }
#pragma unroll
  for (int e = 0; e < E_NUM; ++e)
#pragma unroll
    for (int off = 16; off > 0; off >>= 1)
      acc[e] += __shfl_xor(acc[e], off, 32);

  if (lane == 0) {
    float b0 = -1e30f, b1v = -1e30f;
    int i0 = 0, i1 = 0;
#pragma unroll
    for (int e = 0; e < E_NUM; ++e) {
      float v = acc[e];
      if (v > b0)       { b1v = b0; i1 = i0; b0 = v; i0 = e; }
      else if (v > b1v) { b1v = v; i1 = e; }
    }
    float p1 = __expf(b1v - b0);                 // p0 = 1
    float inv = 1.0f / (1.0f + p1);
    int pos0 = atomicAdd(&cnt[i0], 1);
    int pos1 = atomicAdd(&cnt[i1], 1);
    idx_list[i0 * N_TOK + pos0] = n;
    idx_list[i1 * N_TOK + pos1] = n;
    tok_epos[2 * n]     = (i0 << 13) | pos0;
    tok_epos[2 * n + 1] = (i1 << 13) | pos1;
    tok_w[2 * n]     = inv;
    tok_w[2 * n + 1] = p1 * inv;
  }
}

// ---------------------------------------------------------------------------
__global__ void moe_prefix(const int* __restrict__ cnt, int* __restrict__ base) {
  if (threadIdx.x == 0) {
    int s = 0;
#pragma unroll
    for (int e = 0; e < E_NUM; ++e) { base[e] = s; s += cnt[e]; }
  }
}

// ---------------------------------------------------------------------------
// Kernel B: per-expert fused FFN on a gathered 64-token tile.
// 16 waves. GEMM1: 4(M)x2(N)x2(Ksplit) -> all 16 waves + LDS reduce.
// GEMM2: 4(M)x4(N), 16 accumulator tiles per wave (64x1024 in registers).
// Stage-batched pipeline: per stage, all WMMAs issue, then all LDS reads for
// the next stage -> overlapping fragment live ranges -> non-zero dscnt waits.
// ---------------------------------------------------------------------------
__global__ void __launch_bounds__(512, 1) moe_ffn(
    const float* __restrict__ x,
    const float* __restrict__ w1, const float* __restrict__ b1,
    const float* __restrict__ w2, const float* __restrict__ b2,
    const int* __restrict__ cnt, const int* __restrict__ base,
    const int* __restrict__ idx_list, float* __restrict__ partial) {
  extern __shared__ char smem[];
  __bf16* xs   = (__bf16*)(smem + XS_OFF);
  __bf16* w1s  = (__bf16*)(smem + W1_OFF);
  __bf16* w2s  = (__bf16*)(smem + W2_OFF);
  __bf16* hs   = (__bf16*)(smem + HS_OFF);
  float*  hr   = (float*)(smem + HR_OFF);
  int*    toks = (int*)(smem + TOK_OFF);

  const int e    = blockIdx.y;
  const int tile = blockIdx.x;
  const int cntE = cnt[e];
  const int row0 = tile * M_TILE;
  if (row0 >= cntE) return;                    // uniform per block
  const int baseE = base[e];
  const int tid   = threadIdx.x;

  if (tid < M_TILE) {
    int r = row0 + tid;
    toks[tid] = (r < cntE) ? idx_list[e * N_TOK + r] : 0;  // pad with token 0
  }
  __syncthreads();

  // ---- gather x rows -> A-fragment-major LDS (float4 loads, b64 LDS stores)
  for (int i = tid; i < M_TILE * (C_DIM / 4); i += 512) {
    int r = i >> 8, c = (i & 255) << 2;
    float4 v = *(const float4*)(x + (size_t)toks[r] * C_DIM + c);
    int kk = c & 31;
    int la = (r & 15) + (((kk >> 3) & 1) << 4);
    int eo = (kk & 7) + ((kk >> 4) << 3);       // 4 consecutive K -> contiguous
    __bf16* dst = xs + ((((r >> 4) << 5) + (c >> 5)) << 9) + (la << 4) + eo;
    union { __bf16 h[4]; unsigned long long u; } pk;
    pk.h[0] = (__bf16)v.x; pk.h[1] = (__bf16)v.y;
    pk.h[2] = (__bf16)v.z; pk.h[3] = (__bf16)v.w;
    *(unsigned long long*)dst = pk.u;
  }

  const int wid  = tid >> 5;
  const int lane = tid & 31;
  const int mg = wid >> 2;        // GEMM2: 0..3 (16-row group)
  const int ng = wid & 3;         // GEMM2: 0..3 (256-col group)
  const int m1 = (wid >> 1) & 3;  // GEMM1 tile row group
  const int n1 = wid & 1;         // GEMM1 tile col group
  const int ks = wid >> 3;        // GEMM1 K-split half

  const float* w1e = w1 + (size_t)e * C_DIM * H_DIM;
  const float* w2e = w2 + (size_t)e * H_DIM * C_DIM;
  const float* b1e = b1 + (size_t)e * H_DIM;
  const float* b2e = b2 + (size_t)e * C_DIM;

  // loop-invariant fragment base pointers
  const __bf16* xbase = xs + (((m1 << 5) + (ks << 4)) << 9);   // 16 A-tiles
  const __bf16* wbase = w1s + (((ks << 5) + n1) << 9);         // stride 2 tiles
  const __bf16* w2t   = w2s + ((ng << 4) << 9);                // 16 B-tiles
  float* hrt = hr + (((m1 << 1) + n1) << 8);

  v8f vzero = {};
  v8f acc[16];
#pragma unroll
  for (int t = 0; t < 16; ++t) acc[t] = vzero;

  __syncthreads();

  for (int h0 = 0; h0 < H_DIM; h0 += KCH) {
    // ---- stage w1[:, h0:h0+32] into B-fragment-major tiles ----
    for (int i = tid; i < C_DIM * (KCH / 4); i += 512) {
      int c = i >> 3, j4 = (i & 7) << 2;        // c = K (row), j4 = n (col)
      float4 v = *(const float4*)(w1e + (size_t)c * H_DIM + h0 + j4);
      int kk = c & 31;
      int eo = (kk & 7) + ((kk >> 4) << 3);
      int lb = (((kk >> 3) & 1) << 4);
      __bf16* tb = w1s + (((((c >> 5) << 1)) + (j4 >> 4)) << 9);
      int j = j4 & 15;
      tb[(j + 0 + lb) * 16 + eo] = (__bf16)v.x;
      tb[(j + 1 + lb) * 16 + eo] = (__bf16)v.y;
      tb[(j + 2 + lb) * 16 + eo] = (__bf16)v.z;
      tb[(j + 3 + lb) * 16 + eo] = (__bf16)v.w;
    }
    // ---- stage w2[h0:h0+32, :] into B-fragment-major tiles ----
    for (int i = tid; i < KCH * (C_DIM / 4); i += 512) {
      int j = i >> 8, c = (i & 255) << 2;       // j = K (row), c = n (col)
      float4 v = *(const float4*)(w2e + (size_t)(h0 + j) * C_DIM + c);
      int eo = (j & 7) + ((j >> 4) << 3);
      int lb = (((j >> 3) & 1) << 4);
      __bf16* tb = w2s + ((c >> 4) << 9);
      int cl = c & 15;
      tb[(cl + 0 + lb) * 16 + eo] = (__bf16)v.x;
      tb[(cl + 1 + lb) * 16 + eo] = (__bf16)v.y;
      tb[(cl + 2 + lb) * 16 + eo] = (__bf16)v.z;
      tb[(cl + 3 + lb) * 16 + eo] = (__bf16)v.w;
    }
    __syncthreads();

    // ---- GEMM1: h = relu(x_tile @ w1chunk + b1), K-split across wave pairs.
    // Stage = 2 WMMAs then 8 ds_reads (fragments for kt+2, kt+3).
    v8f c1 = vzero;
    {
      v16bf a0 = load_frag(xbase, lane);
      v16bf b0 = load_frag(wbase, lane);
      v16bf a1 = load_frag(xbase + (1 << 9), lane);
      v16bf b1f = load_frag(wbase + (1 << 10), lane);
#pragma unroll
      for (int kt = 0; kt < 16; kt += 2) {
        c1 = __builtin_amdgcn_wmma_f32_16x16x32_bf16(
                 false, a0, false, b0, (short)0, c1, false, false);
        c1 = __builtin_amdgcn_wmma_f32_16x16x32_bf16(
                 false, a1, false, b1f, (short)0, c1, false, false);
        if (kt + 2 < 16) {
          a0 = load_frag(xbase + ((kt + 2) << 9), lane);
          b0 = load_frag(wbase + ((kt + 2) << 10), lane);
          a1 = load_frag(xbase + ((kt + 3) << 9), lane);
          b1f = load_frag(wbase + ((kt + 3) << 10), lane);
        }
        SGB_WMMA(2); SGB_DSR(8);
      }
    }
    if (ks) {                                   // upper half stores partial
#pragma unroll
      for (int i = 0; i < 8; ++i) hrt[(i << 5) + lane] = c1[i];
    }
    __syncthreads();
    if (!ks) {                                  // lower half reduces + epilogue
      int K  = (n1 << 4) + (lane & 15);
      float bias = b1e[h0 + K];
      int kb = (((K >> 3) & 1) << 4);
      int eo = (K & 7) + ((K >> 4) << 3);
      __bf16* hst = hs + (m1 << 9);
#pragma unroll
      for (int i = 0; i < 8; ++i) {
        int rr = i + ((lane >> 4) << 3);
        float v = c1[i] + hrt[(i << 5) + lane] + bias;
        hst[(rr + kb) * 16 + eo] = (__bf16)fmaxf(v, 0.f);
      }
    }
    __syncthreads();

    // ---- GEMM2: acc += h_chunk[64x32] @ w2chunk[32x1024].
    // Stage = 4 WMMAs then 8 ds_reads (B tiles t+4..t+7): forces 4 live
    // buffers -> waits become dscnt<=6 instead of 0.
    {
      v16bf a2 = load_frag(hs + (mg << 9), lane);
      v16bf bb0 = load_frag(w2t, lane);
      v16bf bb1 = load_frag(w2t + (1 << 9), lane);
      v16bf bb2 = load_frag(w2t + (2 << 9), lane);
      v16bf bb3 = load_frag(w2t + (3 << 9), lane);
#pragma unroll
      for (int t = 0; t < 16; t += 4) {
        acc[t] = __builtin_amdgcn_wmma_f32_16x16x32_bf16(
                     false, a2, false, bb0, (short)0, acc[t], false, false);
        acc[t + 1] = __builtin_amdgcn_wmma_f32_16x16x32_bf16(
                     false, a2, false, bb1, (short)0, acc[t + 1], false, false);
        acc[t + 2] = __builtin_amdgcn_wmma_f32_16x16x32_bf16(
                     false, a2, false, bb2, (short)0, acc[t + 2], false, false);
        acc[t + 3] = __builtin_amdgcn_wmma_f32_16x16x32_bf16(
                     false, a2, false, bb3, (short)0, acc[t + 3], false, false);
        if (t + 4 < 16) {
          bb0 = load_frag(w2t + ((t + 4) << 9), lane);
          bb1 = load_frag(w2t + ((t + 5) << 9), lane);
          bb2 = load_frag(w2t + ((t + 6) << 9), lane);
          bb3 = load_frag(w2t + ((t + 7) << 9), lane);
        }
        SGB_WMMA(4); SGB_DSR(8);
      }
    }
    __syncthreads();
  }

  // ---- epilogue: partial[slot] = acc + b2 (fp32, no atomics) ----
  {
    int hsel = lane >> 4, c0 = lane & 15;
#pragma unroll
    for (int t = 0; t < 16; ++t) {
      int colg = (ng << 8) + (t << 4) + c0;
      float bias2 = b2e[colg];
#pragma unroll
      for (int i = 0; i < 8; ++i) {
        int row  = (mg << 4) + i + (hsel << 3);
        int grow = row0 + row;
        if (grow < cntE)
          partial[(size_t)(baseE + grow) * C_DIM + colg] = acc[t][i] + bias2;
      }
    }
  }
}

// ---------------------------------------------------------------------------
// Kernel C: combine the two weighted partial rows per token (float4).
// ---------------------------------------------------------------------------
__global__ void __launch_bounds__(256) moe_combine(
    const int* __restrict__ tok_epos, const float* __restrict__ tok_w,
    const int* __restrict__ base, const float* __restrict__ partial,
    float* __restrict__ out) {
  int n = blockIdx.x;
  int ep0 = tok_epos[2 * n], ep1 = tok_epos[2 * n + 1];
  float wg0 = tok_w[2 * n], wg1 = tok_w[2 * n + 1];
  const float4* p0 = (const float4*)(partial +
      (size_t)(base[ep0 >> 13] + (ep0 & 8191)) * C_DIM);
  const float4* p1 = (const float4*)(partial +
      (size_t)(base[ep1 >> 13] + (ep1 & 8191)) * C_DIM);
  float4* o = (float4*)(out + (size_t)n * C_DIM);
  for (int c = threadIdx.x; c < C_DIM / 4; c += blockDim.x) {
    float4 a = p0[c], b = p1[c], r;
    r.x = wg0 * a.x + wg1 * b.x;
    r.y = wg0 * a.y + wg1 * b.y;
    r.z = wg0 * a.z + wg1 * b.z;
    r.w = wg0 * a.w + wg1 * b.w;
    o[c] = r;
  }
}

// ---------------------------------------------------------------------------
extern "C" void kernel_launch(void* const* d_in, const int* in_sizes, int n_in,
                              void* d_out, int out_size, void* d_ws, size_t ws_size,
                              hipStream_t stream) {
  const float* x      = (const float*)d_in[0];
  const float* gate_w = (const float*)d_in[1];
  const float* w1     = (const float*)d_in[2];
  const float* b1     = (const float*)d_in[3];
  const float* w2     = (const float*)d_in[4];
  const float* b2     = (const float*)d_in[5];
  float* out = (float*)d_out;

  char* ws = (char*)d_ws;
  int*   cnt      = (int*)(ws + WS_CNT);
  int*   base     = (int*)(ws + WS_BASE);
  int*   idx_list = (int*)(ws + WS_IDX);
  int*   tok_epos = (int*)(ws + WS_EPOS);
  float* tok_w    = (float*)(ws + WS_TOKW);
  float* partial  = (float*)(ws + WS_PART);

  hipMemsetAsync(cnt, 0, E_NUM * sizeof(int), stream);

  moe_gate<<<N_TOK / 8, 256, 0, stream>>>(x, gate_w, cnt, idx_list, tok_epos, tok_w);
  moe_prefix<<<1, 32, 0, stream>>>(cnt, base);

  hipFuncSetAttribute((const void*)moe_ffn,
                      hipFuncAttributeMaxDynamicSharedMemorySize, SMEM_FFN);
  moe_ffn<<<dim3(N_TOK / M_TILE, E_NUM), 512, SMEM_FFN, stream>>>(
      x, w1, b1, w2, b2, cnt, base, idx_list, partial);

  moe_combine<<<N_TOK, 256, 0, stream>>>(tok_epos, tok_w, base, partial, out);
}
